// SelfAttention_25924422599213
// MI455X (gfx1250) — matrix operand order repaired and checked by reading, not compiled, not run on hardware
//
#include <hip/hip_runtime.h>

// ---------------------------------------------------------------------------
// Self-attention forward for MI455X (gfx1250), f16 WMMA everywhere.
// B=2 N=2048 KD=VD=ID=1024 H=16 HD=64.  All GEMMs + attention use
// v_wmma_f32_16x16x32_f16 (f32 accumulate).  Flash-attention keeps the
// 2048x2048 score matrices in registers/LDS only.
// ---------------------------------------------------------------------------

constexpr int Bc  = 2;
constexpr int Nc  = 2048;
constexpr int KDc = 1024;
constexpr int VDc = 1024;
constexpr int IDc = 1024;
constexpr int Hc  = 16;
constexpr int HDc = 64;
constexpr int BNc = Bc * Nc;          // 4096 rows
constexpr float SCALEc = 0.03125f;    // 1024^-0.5
constexpr float EPSc   = 1e-5f;

typedef _Float16 f16;
typedef _Float16 v16h __attribute__((ext_vector_type(16)));
typedef _Float16 v8h  __attribute__((ext_vector_type(8)));
typedef float    v8f  __attribute__((ext_vector_type(8)));

static __device__ __forceinline__ v8f zero8() {
  v8f z;
#pragma unroll
  for (int i = 0; i < 8; ++i) z[i] = 0.0f;
  return z;
}

// A-matrix 16x32 f16 fragment (ISA 7.12.2): lane half offset kb in {0,8};
// halves 0..7 = K kb+0..7, halves 8..15 = K kb+16..23.  Two 16B loads.
static __device__ __forceinline__ v16h load_a_frag(const f16* p) {
  v8h lo = *(const v8h*)(p);
  v8h hi = *(const v8h*)(p + 16);
  v16h r;
#pragma unroll
  for (int i = 0; i < 8; ++i) { r[i] = lo[i]; r[i + 8] = hi[i]; }
  return r;
}

static __device__ __forceinline__ v8f wmma_f16(v16h a, v16h b, v8f c) {
  return __builtin_amdgcn_wmma_f32_16x16x32_f16(
      false, a, false, b, (short)0, c, false, false);
}

static __device__ __forceinline__ float wave_sum(float v) {
#pragma unroll
  for (int m = 16; m >= 1; m >>= 1) v += __shfl_xor(v, m, 32);
  return v;
}

// ---------------------------------------------------------------------------
// LayerNorm a row of length D (f32 in) -> f16 out, same layout.
// ---------------------------------------------------------------------------
__global__ __launch_bounds__(256) void ln_rows_f16(
    const float* __restrict__ X, const float* __restrict__ g,
    const float* __restrict__ bt, f16* __restrict__ out, int D)
{
  const int tid = threadIdx.x;
  const size_t row = blockIdx.x;
  X += row * (size_t)D; out += row * (size_t)D;
  float s = 0.f, s2 = 0.f;
  for (int c = tid; c < D; c += 256) { float v = X[c]; s += v; s2 += v * v; }
  __shared__ float red[2][8];
  float w1 = wave_sum(s), w2 = wave_sum(s2);
  if ((tid & 31) == 0) { red[0][tid >> 5] = w1; red[1][tid >> 5] = w2; }
  __syncthreads();
  if (tid == 0) {
    float t1 = 0.f, t2 = 0.f;
    for (int i = 0; i < 8; ++i) { t1 += red[0][i]; t2 += red[1][i]; }
    red[0][0] = t1; red[1][0] = t2;
  }
  __syncthreads();
  float mu = red[0][0] / D;
  float var = red[1][0] / D - mu * mu;
  float rstd = rsqrtf(var + EPSc);
  for (int c = tid; c < D; c += 256)
    out[c] = (f16)((X[c] - mu) * rstd * g[c] + bt[c]);
}

// ---------------------------------------------------------------------------
// LayerNorm row of length ID taken from a wider matrix (stride ldx), scale by
// oscale, write f16 into head-split layout (B,H,N,HD).
// ---------------------------------------------------------------------------
__global__ __launch_bounds__(256) void ln_split_heads(
    const float* __restrict__ X, int ldx,
    const float* __restrict__ g, const float* __restrict__ bt,
    f16* __restrict__ out, float oscale)
{
  const int tid = threadIdx.x;
  const size_t r = blockIdx.x;                 // b*N + n
  const int bb = (int)(r / Nc), nn = (int)(r % Nc);
  const float* xr = X + r * (size_t)ldx;
  float s = 0.f, s2 = 0.f;
  for (int c = tid; c < IDc; c += 256) { float v = xr[c]; s += v; s2 += v * v; }
  __shared__ float red[2][8];
  float w1 = wave_sum(s), w2 = wave_sum(s2);
  if ((tid & 31) == 0) { red[0][tid >> 5] = w1; red[1][tid >> 5] = w2; }
  __syncthreads();
  if (tid == 0) {
    float t1 = 0.f, t2 = 0.f;
    for (int i = 0; i < 8; ++i) { t1 += red[0][i]; t2 += red[1][i]; }
    red[0][0] = t1; red[1][0] = t2;
  }
  __syncthreads();
  float mu = red[0][0] / IDc;
  float var = red[1][0] / IDc - mu * mu;
  float rstd = rsqrtf(var + EPSc);
  for (int c = tid; c < IDc; c += 256) {
    int h = c >> 6, d = c & 63;
    float v = ((xr[c] - mu) * rstd * g[c] + bt[c]) * oscale;
    out[(((size_t)(bb * Hc + h) * Nc) + nn) * HDc + d] = (f16)v;
  }
}

// ---------------------------------------------------------------------------
// Weight (K,Nw) f32 row-major -> (Nw,K) f16 (so GEMM B-fragments are
// contiguous 32B loads).
// ---------------------------------------------------------------------------
__global__ void transpose_f32_to_f16(const float* __restrict__ W,
                                     f16* __restrict__ Wt, int K, int Nw)
{
  size_t idx = (size_t)blockIdx.x * blockDim.x + threadIdx.x;
  if (idx >= (size_t)K * Nw) return;
  int n = (int)(idx % Nw);
  int k = (int)(idx / Nw);
  Wt[(size_t)n * K + k] = (f16)W[idx];
}

// v raw (B,N,ID) f32 -> (B,H,HD,N) f16 (d-major so P@V B-frags are contiguous)
__global__ void scatter_v_transpose(const float* __restrict__ vraw,
                                    f16* __restrict__ vT)
{
  size_t idx = (size_t)blockIdx.x * blockDim.x + threadIdx.x;
  if (idx >= (size_t)BNc * IDc) return;
  int col = (int)(idx & (IDc - 1));
  size_t row = idx >> 10;
  int bb = (int)(row >> 11);                   // /Nc
  int nn = (int)(row & (Nc - 1));
  int h = col >> 6, d = col & 63;
  vT[(((size_t)(bb * Hc + h) * HDc) + d) * Nc + nn] = (f16)vraw[idx];
}

// ---------------------------------------------------------------------------
// WMMA GEMM: C(M,Nw) f32 = A16(M,K) @ Bt16(Nw,K)^T (+bias).
// Block 256 threads = 8 waves (4x2); wave tile 32x64 (2x4 wmma subtiles);
// block tile 128x128.  All dims multiples of 128 -> no bounds checks.
// ---------------------------------------------------------------------------
__global__ __launch_bounds__(256) void gemm_f16f32(
    const f16* __restrict__ A, const f16* __restrict__ Bt,
    float* __restrict__ C, const float* __restrict__ bias,
    int M, int Nw, int K)
{
  const int tid = threadIdx.x;
  const int lane = tid & 31;
  const int wv = tid >> 5;
  const int waveM = wv >> 1;                   // 0..3
  const int waveN = wv & 1;                    // 0..1
  const int m0 = blockIdx.y * 128 + waveM * 32;
  const int n0 = blockIdx.x * 128 + waveN * 64;
  const int mr = lane & 15;
  const int akb = (lane < 16) ? 0 : 8;         // A half-lane K offset
  const int bkb = (lane < 16) ? 0 : 8;         // B half-lane K offset

  v8f acc[2][4];
#pragma unroll
  for (int i = 0; i < 2; ++i)
#pragma unroll
    for (int j = 0; j < 4; ++j) acc[i][j] = zero8();

  for (int k0 = 0; k0 < K; k0 += 32) {
    v16h a[2], b[4];
#pragma unroll
    for (int i = 0; i < 2; ++i)
      a[i] = load_a_frag(A + (size_t)(m0 + i * 16 + mr) * K + k0 + akb);
#pragma unroll
    for (int j = 0; j < 4; ++j)
      b[j] = load_a_frag(Bt + (size_t)(n0 + j * 16 + mr) * K + k0 + bkb);
#pragma unroll
    for (int i = 0; i < 2; ++i)
#pragma unroll
      for (int j = 0; j < 4; ++j)
        acc[i][j] = wmma_f16(a[i], b[j], acc[i][j]);
  }

  const int erow = (lane < 16) ? 0 : 8;
#pragma unroll
  for (int i = 0; i < 2; ++i) {
#pragma unroll
    for (int j = 0; j < 4; ++j) {
      const int col = n0 + j * 16 + mr;
      const float bv = bias ? bias[col] : 0.0f;
#pragma unroll
      for (int e = 0; e < 8; ++e) {
        const int row = m0 + i * 16 + erow + e;
        C[(size_t)row * Nw + col] = acc[i][j][e] + bv;
      }
    }
  }
}

// ---------------------------------------------------------------------------
// Flash attention, one wave owns 16 query rows, keys processed 32 at a time.
// Q (B,H,N,64) f16 (pre-scaled), K (B,H,N,64) f16, V^T (B,H,64,N) f16.
// Out f16 (B,N,ID).
// ---------------------------------------------------------------------------
__global__ __launch_bounds__(128) void flash_attn(
    const f16* __restrict__ Qg, const f16* __restrict__ Kg,
    const f16* __restrict__ Vg, f16* __restrict__ Og)
{
  __shared__ __align__(16) f16 plds[4][16 * 40];   // per-wave P staging
  const int tid = threadIdx.x;
  const int lane = tid & 31;
  const int wv = tid >> 5;
  const int bh = blockIdx.y;
  const int bb = bh >> 4;
  const int h  = bh & 15;
  const int q0 = blockIdx.x * 64 + wv * 16;
  const int mr = lane & 15;
  const int akb = (lane < 16) ? 0 : 8;
  const int bkb = (lane < 16) ? 0 : 8;

  const f16* Q  = Qg + (size_t)bh * Nc * HDc;
  const f16* Kk = Kg + (size_t)bh * Nc * HDc;
  const f16* V  = Vg + (size_t)bh * HDc * Nc;

  v16h qf[2];
#pragma unroll
  for (int s = 0; s < 2; ++s)
    qf[s] = load_a_frag(Q + (size_t)(q0 + mr) * HDc + s * 32 + akb);

  float mstat[8], lstat[8];
  v8f acc[4];
#pragma unroll
  for (int e = 0; e < 8; ++e) { mstat[e] = -3.0e38f; lstat[e] = 0.0f; }
#pragma unroll
  for (int t = 0; t < 4; ++t) acc[t] = zero8();

  f16* myp = plds[wv];
  const int prow = (lane < 16) ? 0 : 8;

  for (int kb0 = 0; kb0 < Nc; kb0 += 32) {
    // S = Q @ K^T for a 16x32 score tile (2 n-subtiles x 2 K-steps)
    v8f s0 = zero8(), s1 = zero8();
#pragma unroll
    for (int s = 0; s < 2; ++s) {
      v16h b0 = load_a_frag(Kk + (size_t)(kb0 + mr)      * HDc + s * 32 + bkb);
      v16h b1 = load_a_frag(Kk + (size_t)(kb0 + 16 + mr) * HDc + s * 32 + bkb);
      s0 = wmma_f16(qf[s], b0, s0);
      s1 = wmma_f16(qf[s], b1, s1);
    }

    // online softmax (SCALE already folded into Q); row stats reduced over
    // each 16-lane half (C-layout: element e -> row, lane&15 -> col).
    float p0[8], p1[8], corr[8];
#pragma unroll
    for (int e = 0; e < 8; ++e) {
      float bm = fmaxf(s0[e], s1[e]);
#pragma unroll
      for (int m = 1; m < 16; m <<= 1) bm = fmaxf(bm, __shfl_xor(bm, m, 32));
      const float mnew = fmaxf(mstat[e], bm);
      corr[e] = __expf(mstat[e] - mnew);
      mstat[e] = mnew;
      p0[e] = __expf(s0[e] - mnew);
      p1[e] = __expf(s1[e] - mnew);
      float rs = p0[e] + p1[e];
#pragma unroll
      for (int m = 1; m < 16; m <<= 1) rs += __shfl_xor(rs, m, 32);
      lstat[e] = lstat[e] * corr[e] + rs;
    }
#pragma unroll
    for (int t = 0; t < 4; ++t)
#pragma unroll
      for (int e = 0; e < 8; ++e) acc[t][e] *= corr[e];

    // C-layout P -> LDS -> A-layout fragment (per-wave region, HW keeps
    // same-wave DS ops in order; wave_barrier pins scheduling).
#pragma unroll
    for (int e = 0; e < 8; ++e) {
      myp[(prow + e) * 40 + mr]      = (f16)p0[e];
      myp[(prow + e) * 40 + 16 + mr] = (f16)p1[e];
    }
    __builtin_amdgcn_wave_barrier();
    v16h pf = load_a_frag(myp + mr * 40 + akb);
    __builtin_amdgcn_wave_barrier();

    // O += P @ V  (V^T rows are contiguous in keys)
#pragma unroll
    for (int t = 0; t < 4; ++t) {
      v16h vf = load_a_frag(V + (size_t)(t * 16 + mr) * Nc + kb0 + bkb);
      acc[t] = wmma_f16(pf, vf, acc[t]);
    }
  }

  // normalize and store into (B,N,ID) f16 for the output GEMM
#pragma unroll
  for (int t = 0; t < 4; ++t) {
#pragma unroll
    for (int e = 0; e < 8; ++e) {
      const int qrow = q0 + prow + e;
      const float v = acc[t][e] / lstat[e];
      Og[((size_t)bb * Nc + qrow) * IDc + h * HDc + t * 16 + mr] = (f16)v;
    }
  }
}

// ---------------------------------------------------------------------------
// Host orchestration.  Workspace layout (phase-aliased, 72 MB total):
//   [0,8)    kn16  -> later attn16
//   [8,16)   vn16
//   [16,20)  wqkT16   [20,22) wvT16   [22,24) woutT16
//   [24,32)  q16      [32,40) k16
//   [40,72)  qkraw f32 -> later vraw f32 [40,56) + vT16 [56,64)
// ---------------------------------------------------------------------------
extern "C" void kernel_launch(void* const* d_in, const int* in_sizes, int n_in,
                              void* d_out, int out_size, void* d_ws, size_t ws_size,
                              hipStream_t stream)
{
  (void)in_sizes; (void)n_in; (void)out_size; (void)ws_size;
  const float* keys   = (const float*)d_in[0];
  const float* values = (const float*)d_in[1];
  const float* qk_g   = (const float*)d_in[2];
  const float* qk_b   = (const float*)d_in[3];
  const float* vn_g   = (const float*)d_in[4];
  const float* vn_b   = (const float*)d_in[5];
  const float* key_g  = (const float*)d_in[6];
  const float* key_b  = (const float*)d_in[7];
  const float* qry_g  = (const float*)d_in[8];
  const float* qry_b  = (const float*)d_in[9];
  const float* w_qk   = (const float*)d_in[10];
  const float* w_v    = (const float*)d_in[11];
  const float* w_out  = (const float*)d_in[12];
  const float* b_out  = (const float*)d_in[13];
  float* out = (float*)d_out;

  const size_t MB = 1024ull * 1024ull;
  char* w = (char*)d_ws;
  f16* kn16    = (f16*)(w + 0);
  f16* attn16  = (f16*)(w + 0);
  f16* vn16    = (f16*)(w + 8 * MB);
  f16* wqkT    = (f16*)(w + 16 * MB);
  f16* wvT     = (f16*)(w + 20 * MB);
  f16* woutT   = (f16*)(w + 22 * MB);
  f16* q16     = (f16*)(w + 24 * MB);
  f16* k16     = (f16*)(w + 32 * MB);
  float* qkraw = (float*)(w + 40 * MB);
  float* vraw  = (float*)(w + 40 * MB);
  f16* vT16    = (f16*)(w + 56 * MB);

  // 1) input layernorms -> f16
  ln_rows_f16<<<BNc, 256, 0, stream>>>(keys,   qk_g, qk_b, kn16, KDc);
  ln_rows_f16<<<BNc, 256, 0, stream>>>(values, vn_g, vn_b, vn16, VDc);

  // 2) weights -> transposed f16
  transpose_f32_to_f16<<<(KDc * 2 * IDc) / 256, 256, 0, stream>>>(w_qk, wqkT, KDc, 2 * IDc);
  transpose_f32_to_f16<<<(VDc * IDc) / 256, 256, 0, stream>>>(w_v, wvT, VDc, IDc);
  transpose_f32_to_f16<<<(IDc * VDc) / 256, 256, 0, stream>>>(w_out, woutT, IDc, VDc);

  // 3) qk = LN(keys) @ w_qk   (4096 x 2048 x 1024)
  gemm_f16f32<<<dim3((2 * IDc) / 128, BNc / 128), 256, 0, stream>>>(
      kn16, wqkT, qkraw, nullptr, BNc, 2 * IDc, KDc);

  // 4) LN(q)*SCALE and LN(k) -> head-split (B,H,N,64) f16
  ln_split_heads<<<BNc, 256, 0, stream>>>(qkraw,        2 * IDc, qry_g, qry_b, q16, SCALEc);
  ln_split_heads<<<BNc, 256, 0, stream>>>(qkraw + IDc,  2 * IDc, key_g, key_b, k16, 1.0f);

  // 5) v = LN(values) @ w_v, then transpose to (B,H,64,N)
  gemm_f16f32<<<dim3(IDc / 128, BNc / 128), 256, 0, stream>>>(
      vn16, wvT, vraw, nullptr, BNc, IDc, VDc);
  scatter_v_transpose<<<(BNc * IDc) / 256, 256, 0, stream>>>(vraw, vT16);

  // 6) flash attention -> (B,N,ID) f16
  flash_attn<<<dim3(Nc / 64, Bc * Hc), 128, 0, stream>>>(q16, k16, vT16, attn16);

  // 7) out = attn @ w_out + b_out
  gemm_f16f32<<<dim3(VDc / 128, BNc / 128), 256, 0, stream>>>(
      attn16, woutT, out, b_out, BNc, VDc, IDc);
}
